// Head_60421599920770
// MI455X (gfx1250) — matrix operand order repaired and checked by reading, not compiled, not run on hardware
//
#include <hip/hip_runtime.h>
#include <hip/hip_bf16.h>
#include <math.h>

typedef _Float16 v16h __attribute__((ext_vector_type(16)));
typedef _Float16 v8h  __attribute__((ext_vector_type(8)));
typedef float    v8f  __attribute__((ext_vector_type(8)));
typedef int      v4i  __attribute__((ext_vector_type(4)));

#define B_   32
#define T_   2048
#define C_   256
#define HS_  64
#define LNEG (-1e9f)

// workspace byte offsets (all f16): Q | K | VT | WT(q,k,v transposed)
// VT layout: [b][feature(64)][token(2048)]  (V transposed per batch)
#define QOFF  (0u)
#define KOFF  (8u  << 20)
#define VTOFF (16u << 20)
#define WTOFF (24u << 20)   // 3 * 64*256 halves = 96 KB

#ifndef __has_builtin
#define __has_builtin(x) 0
#endif
#if __has_builtin(__builtin_amdgcn_global_load_async_to_lds_b128)
#define ASYNC_LDS 1
#endif

// 16-byte global -> LDS copy: async (ASYNCcnt) when available, else sync.
__device__ __forceinline__ void cp16(const _Float16* g, _Float16* l) {
#ifdef ASYNC_LDS
    __builtin_amdgcn_global_load_async_to_lds_b128(
        (__attribute__((address_space(1))) v4i*)g,
        (__attribute__((address_space(3))) v4i*)l, 0, 0);
#else
    *(v8h*)l = *(const v8h*)g;
#endif
}
__device__ __forceinline__ void cp_wait() {
#ifdef ASYNC_LDS
  #if __has_builtin(__builtin_amdgcn_s_wait_asynccnt)
    __builtin_amdgcn_s_wait_asynccnt(0);
  #else
    asm volatile("s_wait_asynccnt 0" ::: "memory");
  #endif
#endif
}

__device__ __forceinline__ v16h ld16(const _Float16* p) {
    v8h a = *(const v8h*)p;
    v8h b = *(const v8h*)(p + 8);
    return __builtin_shufflevector(a, b, 0,1,2,3,4,5,6,7,8,9,10,11,12,13,14,15);
}
// A-matrix (16x32 f16) fragment: rowbase points at this lane's row, k-offset 0
__device__ __forceinline__ v16h ldA(const _Float16* rowbase, int k0, int hi8) {
    v8h a = *(const v8h*)(rowbase + k0 + 2 * hi8);        // K = k0+2*hi8 .. +7
    v8h b = *(const v8h*)(rowbase + k0 + 2 * hi8 + 8);   // K = k0+2*hi8+8 .. +15
    return __builtin_shufflevector(a, b, 0,1,2,3,4,5,6,7,8,9,10,11,12,13,14,15);
}
__device__ __forceinline__ v8f wmma(v16h a, v16h b, v8f c) {
    return __builtin_amdgcn_wmma_f32_16x16x32_f16(false, a, false, b, (short)0, c, false, false);
}
__device__ __forceinline__ float rmax16(float v) {
    v = fmaxf(v, __shfl_xor(v, 1, 32));
    v = fmaxf(v, __shfl_xor(v, 2, 32));
    v = fmaxf(v, __shfl_xor(v, 4, 32));
    v = fmaxf(v, __shfl_xor(v, 8, 32));
    return v;
}
__device__ __forceinline__ float rsum16(float v) {
    v += __shfl_xor(v, 1, 32);
    v += __shfl_xor(v, 2, 32);
    v += __shfl_xor(v, 4, 32);
    v += __shfl_xor(v, 8, 32);
    return v;
}

// ---------------- K0: W (256x64 fp32) -> WT (64x256 f16), x3 ----------------
__global__ void wtrans_kernel(const float* __restrict__ Wq,
                              const float* __restrict__ Wk,
                              const float* __restrict__ Wv,
                              _Float16* __restrict__ wt) {
    int idx = blockIdx.x * 256 + threadIdx.x;      // 3*256*64 = 49152 total
    if (idx >= 3 * C_ * HS_) return;
    int mat = idx >> 14;            // /16384
    int rem = idx & 16383;          // k*64+n (coalesced read)
    int k = rem >> 6, n = rem & 63;
    const float* W = (mat == 0) ? Wq : (mat == 1) ? Wk : Wv;
    wt[mat * (HS_ * C_) + n * C_ + k] = (_Float16)W[rem];
}

// ---------------- K1: QKV projection via WMMA ----------------
// block: 128 threads (4 waves); each block = one 16-row tile of X.
// wave w computes 16x16 output tile at columns [16w,16w+16) for Q, K and V.
// V is written TRANSPOSED to workspace ([b][f][t]) via an LDS round trip so
// K2 never has to transpose in its hot loop.
__global__ void __launch_bounds__(128)
qkv_kernel(const float* __restrict__ X, const _Float16* __restrict__ wt,
           _Float16* __restrict__ Q, _Float16* __restrict__ K,
           _Float16* __restrict__ VT) {
    __shared__ _Float16 ldsX[16 * 264];            // padded row stride
    __shared__ _Float16 ldsV[16 * 72];             // V tile for transpose
    const int tid  = threadIdx.x;
    const int r0   = blockIdx.x * 16;              // global token tile
    // stage X tile (16x256 fp32 -> f16), coalesced
    #pragma unroll
    for (int j = 0; j < 32; ++j) {
        int idx = j * 128 + tid;                   // 0..4095
        int row = idx >> 8, col = idx & 255;
        ldsX[row * 264 + col] = (_Float16)X[(size_t)(r0 + row) * C_ + col];
    }
    __syncthreads();

    const int wave = tid >> 5, lane = tid & 31;
    const int nl = lane & 15;
    const int hi8  = (lane & 16) ? 8  : 0;
    const int hi16 = (lane & 16) ? 16 : 0;
    const _Float16* arow = &ldsX[nl * 264];
    const _Float16* bq = wt + 0 * (HS_ * C_) + (wave * 16 + nl) * C_;
    const _Float16* bk = wt + 1 * (HS_ * C_) + (wave * 16 + nl) * C_;
    const _Float16* bv = wt + 2 * (HS_ * C_) + (wave * 16 + nl) * C_;

    v8f aq = {}, ak = {}, av = {};
    #pragma unroll
    for (int k0 = 0; k0 < C_; k0 += 32) {
        v16h A  = ldA(arow, k0, hi8);
        v16h Bq = ld16(bq + k0 + hi16);
        v16h Bk = ld16(bk + k0 + hi16);
        v16h Bv = ld16(bv + k0 + hi16);
        aq = wmma(A, Bq, aq);
        ak = wmma(A, Bk, ak);
        av = wmma(A, Bv, av);
    }
    const int col = wave * 16 + nl;
    #pragma unroll
    for (int r = 0; r < 8; ++r) {
        size_t g = (size_t)(r0 + r + hi8) * HS_ + col;
        Q[g] = (_Float16)aq[r];
        K[g] = (_Float16)ak[r];
        ldsV[(r + hi8) * 72 + col] = (_Float16)av[r];   // stage V for transpose
    }
    __syncthreads();
    // cooperative transposed store: VT[(b*64+f)*2048 + t], 16B vectors
    {
        const int f = tid >> 1, tseg = (tid & 1) * 8;
        v8h o;
        #pragma unroll
        for (int j = 0; j < 8; ++j) o[j] = ldsV[(tseg + j) * 72 + f];
        const int b = r0 >> 11, tloc = (r0 & (T_ - 1)) + tseg;
        *(v8h*)&VT[((size_t)b * HS_ + f) * T_ + tloc] = o;
    }
}

// ---------------- K2: flash attention ----------------
// 128 threads (4 waves) cover 64 consecutive queries; waves share one K/VT
// LDS tile per 32-key step. Double-buffered: tile kt+1 is staged with
// GLOBAL_LOAD_ASYNC_TO_LDS_B128 while tile kt is computed. All waves run the
// block's max causal tile count; fully-masked tiles contribute exp(-1e9-m)==0,
// so no divergence around barriers/WMMA (EXEC all-1s).
__global__ void __launch_bounds__(128)
attn_kernel(const _Float16* __restrict__ Q, const _Float16* __restrict__ Kg,
            const _Float16* __restrict__ VTg, const unsigned char* __restrict__ amask,
            float* __restrict__ out) {
    __shared__ _Float16 ldsK[2 * 32 * 72];    // [buf][key_local][feature]
    __shared__ _Float16 ldsVT[2 * 64 * 40];   // [buf][feature][key_local]
    __shared__ _Float16 ldsP[4 * 16 * 40];    // per-wave [q_row][key_local]

    const int tid  = threadIdx.x;
    const int wave = tid >> 5, lane = tid & 31;
    const int nl   = lane & 15;
    const int hi8  = (lane & 16) ? 8  : 0;
    const int hi16 = (lane & 16) ? 16 : 0;
    const int b   = blockIdx.x >> 5;          // T/64 = 32 blocks per batch
    const int qtb = blockIdx.x & 31;
    const int q0  = qtb * 64 + wave * 16;     // this wave's query tile
    const size_t batch_base = (size_t)b * T_;
    const _Float16* vtb = VTg + (size_t)b * HS_ * T_;
    _Float16* myP = ldsP + wave * (16 * 40);

    // staging roles
    const int krow = tid >> 2, kquad = (tid & 3) * 16;   // K: 32 rows x 4 chunks
    const int vrow = tid >> 1, vseg  = (tid & 1) * 16;   // VT: 64 rows x 2 chunks

    // Q A-fragments (K-chunks c=0,1 over HS=64)
    const _Float16* qrow = Q + (batch_base + q0 + nl) * HS_;
    v16h Qa = ldA(qrow, 0, hi8);
    v16h Qb = ldA(qrow, 32, hi8);

    v8f acc0 = {}, acc1 = {}, acc2 = {}, acc3 = {};
    float m[8], ll[8];
    #pragma unroll
    for (int r = 0; r < 8; ++r) { m[r] = -INFINITY; ll[r] = 0.0f; }

    const int nkt = 2 * qtb + 2;              // covers keys 0 .. qtb*64+63

    // prologue: stage tile 0 into buffer 0
    {
        const _Float16* ks = Kg + (batch_base + krow) * HS_ + kquad;
        cp16(ks,     &ldsK[krow * 72 + kquad]);
        cp16(ks + 8, &ldsK[krow * 72 + kquad + 8]);
        const _Float16* vs = vtb + (size_t)vrow * T_ + vseg;
        cp16(vs,     &ldsVT[vrow * 40 + vseg]);
        cp16(vs + 8, &ldsVT[vrow * 40 + vseg + 8]);
    }

    for (int kt = 0; kt < nkt; ++kt) {
        const int kc = kt * 32;
        cp_wait();                // current buffer's async copies done
        __syncthreads();          // ... and visible to all waves

        const int cur = kt & 1, nxt = cur ^ 1;
        const _Float16* kbuf = &ldsK[cur * (32 * 72)];
        const _Float16* vbuf = &ldsVT[cur * (64 * 40)];

        if (kt + 1 < nkt) {       // stage next tile while computing this one
            const _Float16* ks = Kg + (batch_base + kc + 32 + krow) * HS_ + kquad;
            cp16(ks,     &ldsK[nxt * (32 * 72) + krow * 72 + kquad]);
            cp16(ks + 8, &ldsK[nxt * (32 * 72) + krow * 72 + kquad + 8]);
            const _Float16* vs = vtb + (size_t)vrow * T_ + kc + 32 + vseg;
            cp16(vs,     &ldsVT[nxt * (64 * 40) + vrow * 40 + vseg]);
            cp16(vs + 8, &ldsVT[nxt * (64 * 40) + vrow * 40 + vseg + 8]);
        }

        // S = Q @ K^T for two 16-key column tiles
        v8f sA = {}, sB = {};
        sA = wmma(Qa, ld16(&kbuf[(0 * 16 + nl) * 72 + 0  + hi16]), sA);
        sA = wmma(Qb, ld16(&kbuf[(0 * 16 + nl) * 72 + 32 + hi16]), sA);
        sB = wmma(Qa, ld16(&kbuf[(1 * 16 + nl) * 72 + 0  + hi16]), sB);
        sB = wmma(Qb, ld16(&kbuf[(1 * 16 + nl) * 72 + 32 + hi16]), sB);

        // scale + causal + padding mask (reference -1e9 semantics)
        const int key0 = kc + nl, key1 = kc + 16 + nl;
        const bool p0ok = amask[batch_base + key0] != 0;
        const bool p1ok = amask[batch_base + key1] != 0;
        float s0[8], s1[8];
        #pragma unroll
        for (int r = 0; r < 8; ++r) {
            int q = q0 + r + hi8;
            float x0 = sA[r] * 0.125f;
            float x1 = sB[r] * 0.125f;
            if (key0 > q || !p0ok) x0 = LNEG;
            if (key1 > q || !p1ok) x1 = LNEG;
            s0[r] = x0; s1[r] = x1;
        }
        // online softmax (per-row stats via half-wave reductions)
        float alpha[8];
        #pragma unroll
        for (int r = 0; r < 8; ++r) {
            float rm = rmax16(fmaxf(s0[r], s1[r]));
            float mn = fmaxf(m[r], rm);
            alpha[r] = __expf(m[r] - mn);
            m[r] = mn;
            float p0 = __expf(s0[r] - mn);
            float p1 = __expf(s1[r] - mn);
            ll[r] = ll[r] * alpha[r] + rsum16(p0 + p1);
            myP[(r + hi8) * 40 + nl]      = (_Float16)p0;
            myP[(r + hi8) * 40 + 16 + nl] = (_Float16)p1;
        }
        #pragma unroll
        for (int r = 0; r < 8; ++r) {
            acc0[r] *= alpha[r]; acc1[r] *= alpha[r];
            acc2[r] *= alpha[r]; acc3[r] *= alpha[r];
        }
        __syncthreads();   // P store (C-layout) -> P reload (A-layout)

        // O += P @ V  (VT in LDS gives contiguous B-frags)
        v16h P = ldA(&myP[nl * 40], 0, hi8);
        acc0 = wmma(P, ld16(&vbuf[(0 * 16 + nl) * 40 + hi16]), acc0);
        acc1 = wmma(P, ld16(&vbuf[(1 * 16 + nl) * 40 + hi16]), acc1);
        acc2 = wmma(P, ld16(&vbuf[(2 * 16 + nl) * 40 + hi16]), acc2);
        acc3 = wmma(P, ld16(&vbuf[(3 * 16 + nl) * 40 + hi16]), acc3);
    }

    // epilogue: normalize and store fp32
    #pragma unroll
    for (int r = 0; r < 8; ++r) {
        size_t g = (batch_base + q0 + r + hi8) * HS_;
        float inv = 1.0f / ll[r];
        out[g + 0 * 16 + nl] = acc0[r] * inv;
        out[g + 1 * 16 + nl] = acc1[r] * inv;
        out[g + 2 * 16 + nl] = acc2[r] * inv;
        out[g + 3 * 16 + nl] = acc3[r] * inv;
    }
}

extern "C" void kernel_launch(void* const* d_in, const int* in_sizes, int n_in,
                              void* d_out, int out_size, void* d_ws, size_t ws_size,
                              hipStream_t stream) {
    (void)in_sizes; (void)n_in; (void)out_size; (void)ws_size;
    const float* X  = (const float*)d_in[0];
    const float* Wq = (const float*)d_in[1];
    const float* Wk = (const float*)d_in[2];
    const float* Wv = (const float*)d_in[3];
    const unsigned char* amask = (const unsigned char*)d_in[4];
    float* out = (float*)d_out;

    _Float16* Qh  = (_Float16*)((char*)d_ws + QOFF);
    _Float16* Kh  = (_Float16*)((char*)d_ws + KOFF);
    _Float16* VTh = (_Float16*)((char*)d_ws + VTOFF);
    _Float16* wt  = (_Float16*)((char*)d_ws + WTOFF);

    wtrans_kernel<<<dim3((3 * C_ * HS_ + 255) / 256), dim3(256), 0, stream>>>(Wq, Wk, Wv, wt);
    qkv_kernel<<<dim3((B_ * T_) / 16), dim3(128), 0, stream>>>(X, wt, Qh, Kh, VTh);
    attn_kernel<<<dim3(B_ * (T_ / 64)), dim3(128), 0, stream>>>(Qh, Kh, VTh, amask, out);
}
